// ContainerHeteroGAT_76897094467749
// MI455X (gfx1250) — matrix operand order repaired and checked by reading, not compiled, not run on hardware
//
#include <hip/hip_runtime.h>

// ---------------- problem constants (from reference) ----------------
#define NN   50000      // nodes
#define NE   800000     // edges per relation
#define NH   4          // heads
#define NC   64         // per-head channels
#define HC   256        // NH*NC
#define NEG_SLOPE 0.2f

#define BM   128        // GEMM rows per block (8 waves x 16)
#define BN   64         // GEMM cols per block (4 wmma subtiles)

typedef _Float16 half_t;
typedef __attribute__((ext_vector_type(8)))  _Float16 v8h;
typedef __attribute__((ext_vector_type(16))) _Float16 v16h;
typedef __attribute__((ext_vector_type(8)))  float    v8f;

// ---------------- helpers ----------------
__device__ __forceinline__ unsigned flipf(float f) {
    unsigned u = __float_as_uint(f);
    return (u & 0x80000000u) ? ~u : (u | 0x80000000u);
}
__device__ __forceinline__ float unflipf(unsigned u) {
    u = (u & 0x80000000u) ? (u & 0x7fffffffu) : ~u;
    return __uint_as_float(u);
}
__device__ __forceinline__ void atomic_add_f32(float* p, float v) {
    __hip_atomic_fetch_add(p, v, __ATOMIC_RELAXED, __HIP_MEMORY_SCOPE_AGENT);
}

// ---------------- elementwise kernels ----------------
__global__ void k_cast_f16(const float* __restrict__ in, half_t* __restrict__ out,
                           long long n, int relu) {
    long long i = (long long)blockIdx.x * blockDim.x + threadIdx.x;
    if (i >= n) return;
    float v = in[i];
    if (relu && v < 0.0f) v = 0.0f;
    out[i] = (half_t)v;
}

__global__ void k_fill_u32(unsigned* __restrict__ p, unsigned v, long long n) {
    long long i = (long long)blockIdx.x * blockDim.x + threadIdx.x;
    if (i >= n) return;
    p[i] = v;
}

// out[n*M + j] = b[0][j] + b[1][j] + b[2][j]   (bias of the 3 relations folded in)
__global__ void k_init_bias3(float* __restrict__ out, const float* __restrict__ b,
                             int M, long long total) {
    long long i = (long long)blockIdx.x * blockDim.x + threadIdx.x;
    if (i >= total) return;
    int j = (int)(i % (long long)M);
    out[i] = b[j] + b[M + j] + b[2 * M + j];
}

// ---------------- WMMA GEMM: C[N x M] = A[N x K](f16) * B[K x M](f16) (+bias) ----
// Block = 8 waves computes a BM x BN (128x64) tile; wave w owns 16 rows x 64 cols
// (4 accumulators -> 4 v_wmma per A-fragment load).
// B tile (32 x 64) is staged through double-buffered LDS, pre-swizzled into WMMA
// fragment order so each lane reads its fragment as 2x ds_load_b128; the next
// tile's coalesced global_load_b128 is issued under the current WMMAs.
// Ragged last row-block is clamped (duplicate tiles recompute identical values),
// so every wave is always fully active: EXEC stays all-ones (WMMA requirement)
// and no exec save/restore appears in the inner loop.
// VGPR layouts per CDNA5 ISA 7.12.2:
//  A 16x32 f16 : lane%16 = row; lanes 0-15 hold K {0..7,16..23}, lanes 16-31 +8
//               -> per lane: two contiguous 16B chunks at half-offsets khalf*8, 16+khalf*8
//  B 32x16 f16 : lane%16 = col; lanes 0-15 hold K {0..7,16..23}, lanes 16-31 +8 (as A)
//  C/D 16x16 f32: VGPR v -> row v + 8*(lane>=16), col = lane%16
__device__ __forceinline__ void stage_B(half_t* __restrict__ lb,
                                        const half_t* __restrict__ B,
                                        int k0, int M, int cbase0, int t) {
    int r  = t >> 3;                // B row within 32-row tile
    int cb = (t & 7) * 8;           // 8 contiguous cols
    const half_t* src = B + (size_t)(k0 + r) * M + cbase0 + cb;
    int kh = (r >> 3) & 1;          // lane-half that holds this row
    int el = (r & 7) + ((r >> 4) << 3); // element index within fragment
#pragma unroll
    for (int q = 0; q < 8; ++q) {
        int c  = cb + q;
        int ct = c >> 4;            // col subtile
        int n  = c & 15;            // col within subtile
        lb[(size_t)((ct * 32) + kh * 16 + n) * 16 + el] = src[q];
    }
}

__global__ __launch_bounds__(256)
void k_gemm_wmma(const half_t* __restrict__ A, const half_t* __restrict__ B,
                 float* __restrict__ C, int Nrows, int K, int M,
                 const float* __restrict__ bias) {
    __shared__ v16h lbv[2][4 * 32];         // 2 x 4KB double buffer

    int wave    = threadIdx.x >> 5;
    int lane    = threadIdx.x & 31;
    int tilesBN = M / BN;                   // M=256 -> 4, M=64 -> 1
    int rowBlk  = blockIdx.x / tilesBN;
    int colBlk  = blockIdx.x - rowBlk * tilesBN;
    int rowBase = rowBlk * BM + wave * 16;
    if (rowBase > Nrows - 16) rowBase = Nrows - 16;   // clamp: all waves active

    int lo     = lane & 15;
    int khalf  = lane >> 4;
    int row    = rowBase + lo;
    int cbase0 = colBlk * BN;

    v8f acc[4] = {{}, {}, {}, {}};

    stage_B((half_t*)lbv[0], B, 0, M, cbase0, threadIdx.x);
    __syncthreads();

    int buf = 0;
    for (int k0 = 0; k0 < K; k0 += 32) {
        if (k0 + 32 < K)
            stage_B((half_t*)lbv[buf ^ 1], B, k0 + 32, M, cbase0, threadIdx.x);

        // A fragment: two contiguous 16B chunks -> 2x global_load_b128
        const half_t* Arow = A + (size_t)row * K + k0 + khalf * 8;
        v8h alo = *(const v8h*)(Arow);
        v8h ahi = *(const v8h*)(Arow + 16);
        v16h a;
#pragma unroll
        for (int j = 0; j < 8; ++j) { a[j] = alo[j]; a[8 + j] = ahi[j]; }

#pragma unroll
        for (int ct = 0; ct < 4; ++ct) {
            v16h b = lbv[buf][ct * 32 + lane];
            acc[ct] = __builtin_amdgcn_wmma_f32_16x16x32_f16(
                false, a, false, b, (short)0, acc[ct], false, false);
        }
        __syncthreads();
        buf ^= 1;
    }

    int mBase = rowBase + khalf * 8;
#pragma unroll
    for (int ct = 0; ct < 4; ++ct) {
        int col = cbase0 + ct * 16 + lo;
        float bv = bias ? bias[col] : 0.0f;
        float* Cc = C + (size_t)mBase * M + col;
#pragma unroll
        for (int v = 0; v < 8; ++v) {
            Cc[(size_t)v * M] = acc[ct][v] + bv;
        }
    }
}

// ---------------- attention: per-(node,head) dot products --------------------
__global__ void k_node_alpha(const float* __restrict__ h,
                             const float* __restrict__ a_s, const float* __restrict__ a_d,
                             float* __restrict__ als, float* __restrict__ ald, int total) {
    int i = blockIdx.x * blockDim.x + threadIdx.x;
    if (i >= total) return;
    int hh = i & (NH - 1);
    int n  = i >> 2;
    const float* hp  = h + (size_t)n * HC + hh * NC;
    const float* asp = a_s + hh * NC;
    const float* adp = a_d + hh * NC;
    float s = 0.0f, d = 0.0f;
#pragma unroll 4
    for (int c = 0; c < NC; ++c) {
        float v = hp[c];
        s += v * asp[c];
        d += v * adp[c];
    }
    als[i] = s;
    ald[i] = d;
}

// ---------------- edge passes ----------------
__device__ __forceinline__ void edge_nodes(const int* __restrict__ e, int E, int idx,
                                           int& src, int& dst) {
    if (idx < E) { src = e[idx]; dst = e[E + idx]; }
    else         { src = dst = idx - E; }        // appended self-loop
}

__global__ void k_edge_max(const int* __restrict__ e, int E, int E_tot,
                           const float* __restrict__ als, const float* __restrict__ ald,
                           unsigned* __restrict__ mbuf) {
    int idx = blockIdx.x * blockDim.x + threadIdx.x;
    if (idx >= E_tot) return;
    int src, dst;
    edge_nodes(e, E, idx, src, dst);
#pragma unroll
    for (int hh = 0; hh < NH; ++hh) {
        float lg = als[src * NH + hh] + ald[dst * NH + hh];
        lg = (lg > 0.0f) ? lg : NEG_SLOPE * lg;          // leaky_relu
        atomicMax(mbuf + dst * NH + hh, flipf(lg));
    }
}

__global__ void k_edge_exp(const int* __restrict__ e, int E, int E_tot,
                           const float* __restrict__ als, const float* __restrict__ ald,
                           const unsigned* __restrict__ mbuf,
                           float* __restrict__ denom, float* __restrict__ exbuf) {
    int idx = blockIdx.x * blockDim.x + threadIdx.x;
    if (idx >= E_tot) return;
    int src, dst;
    edge_nodes(e, E, idx, src, dst);
#pragma unroll
    for (int hh = 0; hh < NH; ++hh) {
        float lg = als[src * NH + hh] + ald[dst * NH + hh];
        lg = (lg > 0.0f) ? lg : NEG_SLOPE * lg;
        float m = unflipf(mbuf[dst * NH + hh]);
        if (!(m > -3.0e38f)) m = 0.0f;                   // where(isfinite(m), m, 0)
        float ex = __expf(lg - m);
        exbuf[(size_t)idx * NH + hh] = ex;
        atomic_add_f32(denom + dst * NH + hh, ex);
    }
}

// one thread per (edge, head, 4-channel chunk): float4 gather of h[src] + 4
// independent global_atomic_add_f32 (h and out are L2-resident: 51MB each < 192MB)
__global__ void k_edge_scatter(const int* __restrict__ e, int E, long long total,
                               const float* __restrict__ h,
                               const float* __restrict__ exbuf,
                               const float* __restrict__ denom,
                               float* __restrict__ out, int concat) {
    long long i = (long long)blockIdx.x * blockDim.x + threadIdx.x;
    if (i >= total) return;
    int rem  = (int)(i & 63);
    int eidx = (int)(i >> 6);
    int hh   = rem >> 4;
    int cq   = (rem & 15) * 4;
    int src, dst;
    edge_nodes(e, E, eidx, src, dst);
    float alpha = exbuf[(size_t)eidx * NH + hh] / (denom[dst * NH + hh] + 1e-16f);
    const float4 hv = *(const float4*)(h + (size_t)src * HC + hh * NC + cq);
    if (concat) {
        float* o = out + (size_t)dst * HC + hh * NC + cq;
        atomic_add_f32(o + 0, alpha * hv.x);
        atomic_add_f32(o + 1, alpha * hv.y);
        atomic_add_f32(o + 2, alpha * hv.z);
        atomic_add_f32(o + 3, alpha * hv.w);
    } else {
        float s = alpha * 0.25f;                         // fold head mean
        float* o = out + (size_t)dst * NC + cq;
        atomic_add_f32(o + 0, s * hv.x);
        atomic_add_f32(o + 1, s * hv.y);
        atomic_add_f32(o + 2, s * hv.z);
        atomic_add_f32(o + 3, s * hv.w);
    }
}

// ---------------- host side ----------------
static inline dim3 grid_for(long long n) { return dim3((unsigned)((n + 255) / 256)); }

static void launch_gemm(hipStream_t s, const half_t* A, const half_t* B, float* C,
                        int Nrows, int K, int M, const float* bias) {
    int tilesBN = M / BN;
    int rowBlks = (Nrows + BM - 1) / BM;
    k_gemm_wmma<<<dim3((unsigned)(rowBlks * tilesBN)), 256, 0, s>>>(A, B, C, Nrows, K, M, bias);
}

static void run_layer(hipStream_t s, const half_t* xh, int K,
                      const int* const* edges, const bool* selfloop,
                      const float* W, const float* as_, const float* ad_,
                      const float* bset, int concat, float* layo,
                      half_t* Wh, float* h, float* als, float* ald,
                      unsigned* mbuf, float* denom, float* exbuf) {
    const int Mout = concat ? HC : NC;
    long long tot = (long long)NN * Mout;
    k_init_bias3<<<grid_for(tot), 256, 0, s>>>(layo, bset, Mout, tot);

    for (int r = 0; r < 3; ++r) {
        long long wn = (long long)K * HC;
        k_cast_f16<<<grid_for(wn), 256, 0, s>>>(W + (size_t)r * K * HC, Wh, wn, 0);

        launch_gemm(s, xh, Wh, h, NN, K, HC, nullptr);

        k_node_alpha<<<grid_for((long long)NN * NH), 256, 0, s>>>(
            h, as_ + r * HC, ad_ + r * HC, als, ald, NN * NH);

        k_fill_u32<<<grid_for((long long)NN * NH), 256, 0, s>>>(mbuf, 0x007FFFFFu, (long long)NN * NH);
        k_fill_u32<<<grid_for((long long)NN * NH), 256, 0, s>>>((unsigned*)denom, 0u, (long long)NN * NH);

        int E_tot = NE + (selfloop[r] ? NN : 0);
        k_edge_max<<<grid_for(E_tot), 256, 0, s>>>(edges[r], NE, E_tot, als, ald, mbuf);
        k_edge_exp<<<grid_for(E_tot), 256, 0, s>>>(edges[r], NE, E_tot, als, ald, mbuf, denom, exbuf);

        long long stot = (long long)E_tot * 64;
        k_edge_scatter<<<grid_for(stot), 256, 0, s>>>(edges[r], NE, stot, h, exbuf, denom, layo, concat);
    }
}

extern "C" void kernel_launch(void* const* d_in, const int* in_sizes, int n_in,
                              void* d_out, int out_size, void* d_ws, size_t ws_size,
                              hipStream_t stream) {
    const float* x        = (const float*)d_in[0];
    const int*   e_blocks = (const int*)d_in[1];
    const int*   e_spat   = (const int*)d_in[2];
    const int*   e_simil  = (const int*)d_in[3];
    const float* W1  = (const float*)d_in[4];
    const float* a1s = (const float*)d_in[5];
    const float* a1d = (const float*)d_in[6];
    const float* b1  = (const float*)d_in[7];
    const float* W2  = (const float*)d_in[8];
    const float* a2s = (const float*)d_in[9];
    const float* a2d = (const float*)d_in[10];
    const float* b2  = (const float*)d_in[11];
    const float* Wl  = (const float*)d_in[12];
    const float* bl  = (const float*)d_in[13];
    float* out = (float*)d_out;

    // carve workspace (~145 MB), 256B-aligned chunks
    char* w = (char*)d_ws;
    auto take = [&](size_t bytes) { char* p = w; w += (bytes + 255) & ~(size_t)255; return p; };
    half_t*   xh    = (half_t*)take((size_t)NN * HC * sizeof(half_t));
    half_t*   Wh    = (half_t*)take((size_t)HC * HC * sizeof(half_t));
    float*    h     = (float*)take((size_t)NN * HC * sizeof(float));
    float*    layo  = (float*)take((size_t)NN * HC * sizeof(float));
    float*    als   = (float*)take((size_t)NN * NH * sizeof(float));
    float*    ald   = (float*)take((size_t)NN * NH * sizeof(float));
    unsigned* mbuf  = (unsigned*)take((size_t)NN * NH * sizeof(unsigned));
    float*    denom = (float*)take((size_t)NN * NH * sizeof(float));
    float*    exbuf = (float*)take((size_t)(NE + NN) * NH * sizeof(float));

    const int* edges[3] = { e_blocks, e_spat, e_simil };
    const bool selfloop[3] = { false, true, true };   // 'blocks' has no self-loops

    // ---- layer 1: [N,128] -> concat [N,256] ----
    k_cast_f16<<<grid_for((long long)NN * 128), 256, 0, stream>>>(x, xh, (long long)NN * 128, 0);
    run_layer(stream, xh, 128, edges, selfloop, W1, a1s, a1d, b1, /*concat=*/1,
              layo, Wh, h, als, ald, mbuf, denom, exbuf);

    // ---- relu + layer 2: [N,256] -> mean [N,64] ----
    k_cast_f16<<<grid_for((long long)NN * HC), 256, 0, stream>>>(layo, xh, (long long)NN * HC, 1);
    run_layer(stream, xh, HC, edges, selfloop, W2, a2s, a2d, b2, /*concat=*/0,
              layo, Wh, h, als, ald, mbuf, denom, exbuf);

    // ---- relu + final linear: [N,64] @ [64,64] + bl -> d_out ----
    k_cast_f16<<<grid_for((long long)NN * NC), 256, 0, stream>>>(layo, xh, (long long)NN * NC, 1);
    k_cast_f16<<<grid_for(64 * 64), 256, 0, stream>>>(Wl, Wh, 64 * 64, 0);
    launch_gemm(stream, xh, Wh, out, NN, 64, 64, bl);
}
